// Qwen3Attention_78993038508230
// MI455X (gfx1250) — matrix operand order repaired and checked by reading, not compiled, not run on hardware
//
#include <hip/hip_runtime.h>

// ---------------------------------------------------------------------------
// Qwen3 attention block for MI455X (gfx1250, wave32, WMMA bf16 + TDM)
// ---------------------------------------------------------------------------

constexpr int BB   = 2;
constexpr int SS   = 2048;
constexpr int HH   = 16;     // query heads
constexpr int KVH  = 8;      // kv heads
constexpr int DD   = 128;    // head dim
constexpr int HIDC = 2048;
constexpr float EPSC   = 1e-6f;
constexpr float SCALEC = 0.08838834764831845f;  // 1/sqrt(128)

typedef __attribute__((ext_vector_type(16))) __bf16          v16bf;
typedef __attribute__((ext_vector_type(8)))  float           v8f;
typedef __attribute__((ext_vector_type(8)))  unsigned short  v8u16;
typedef __attribute__((ext_vector_type(4)))  unsigned int    v4u;
typedef __attribute__((ext_vector_type(8)))  int             v8i;
typedef __attribute__((ext_vector_type(4)))  int             v4i;

#if defined(__has_builtin)
#if __has_builtin(__builtin_amdgcn_tensor_load_to_lds) && \
    __has_builtin(__builtin_amdgcn_s_wait_tensorcnt)
#define USE_TDM 1
#endif
#endif

union Frag {
    v16bf          bf;
    v8u16          h[2];
    unsigned short us[16];
};

__device__ __forceinline__ unsigned short f2bf(float f) {
    unsigned int u = __builtin_bit_cast(unsigned int, f);
    unsigned int r = u + 0x7FFFu + ((u >> 16) & 1u);   // round-to-nearest-even
    return (unsigned short)(r >> 16);
}

__device__ __forceinline__ v8f vzero8() {
    v8f z;
#pragma unroll
    for (int i = 0; i < 8; ++i) z[i] = 0.0f;
    return z;
}

__device__ __forceinline__ v8f wmma_bf16(v16bf a, v16bf b, v8f c) {
    // D(16x16 f32) = A(16x32 bf16) * B(32x16 bf16) + C
    return __builtin_amdgcn_wmma_f32_16x16x32_bf16(
        /*neg_a=*/false, a, /*neg_b=*/false, b,
        /*c_mod=*/(short)0, c, /*reuse_a=*/false, /*reuse_b=*/false);
}

// ---------------------------------------------------------------------------
// f32 -> bf16 bulk conversion (b128 loads, b64 stores); n4 = count/4
// ---------------------------------------------------------------------------
__global__ __launch_bounds__(256) void cvt_bf16_kernel(
    const float* __restrict__ src, unsigned short* __restrict__ dst, int n4)
{
    const int stride = gridDim.x * blockDim.x;
    for (int i = blockIdx.x * blockDim.x + threadIdx.x; i < n4; i += stride) {
        const float4 f = ((const float4*)src)[i];
        ushort4 o;
        o.x = f2bf(f.x); o.y = f2bf(f.y); o.z = f2bf(f.z); o.w = f2bf(f.w);
        ((ushort4*)dst)[i] = o;
    }
}

// ---------------------------------------------------------------------------
// bf16 WMMA GEMM: C(f32, MxN) = A(MxK) * B(KxN); A/B row-major bf16 (u16).
// Block = 256 threads = 8 waves; macro tile 64(M) x 64(N); K step 64.
// A tile staged by the Tensor Data Mover (pad fields reproduce the 72-u16
// bank-padded LDS stride); B tile staged transposed by vector loads.
// ---------------------------------------------------------------------------
constexpr int AST = 72;   // a_s row stride (u16): 64 data + 4-DWORD TDM pad
constexpr int BST = 80;   // b_s row stride (u16), 160B: 32B-aligned fragments

__global__ __launch_bounds__(256) void gemm_wmma_kernel(
    const unsigned short* __restrict__ A, const unsigned short* __restrict__ Bm,
    float* __restrict__ C, int M, int N, int K)
{
    __shared__ alignas(32) unsigned short a_s[64 * AST];  // [row][k]
    __shared__ alignas(32) unsigned short b_s[64 * BST];  // [n][k] (transposed)

    const int tid   = threadIdx.x;
    const int lane  = tid & 31;
    const int w     = tid >> 5;
    const int wr    = w >> 1;          // 0..3 : 16-row sub-tile
    const int wc2   = (w & 1) * 2;     // 0/2  : first of two 16-col sub-tiles
    const int m0    = blockIdx.y * 64;
    const int n0    = blockIdx.x * 64;
    const int off   = (lane >> 4) * 16;  // A-frag K offset per lane half
    const int halfk = (lane >> 4) * 16;  // B-frag K offset per lane half

    v8f c0 = vzero8(), c1 = vzero8();

    for (int k0 = 0; k0 < K; k0 += 64) {
#if defined(USE_TDM)
        // ---- A tile (64 rows x 64 k) via Tensor Data Mover, wave 0 issues ----
        if (tid < 32) {
            const unsigned long long ga =
                (unsigned long long)(const void*)(A + (size_t)m0 * K + k0);
            const unsigned int lbase =
                (unsigned int)(unsigned long long)(const void*)&a_s[0];
            v4u g0;
            g0[0] = 1u;                                      // count=1 (valid D#)
            g0[1] = lbase;                                   // lds_addr
            g0[2] = (unsigned int)ga;                        // global_addr[31:0]
            g0[3] = ((unsigned int)(ga >> 32) & 0x01FFFFFFu) // global_addr[56:32]
                    | 0x80000000u;                           // type=2 (image)
            v8i g1;
            // data_size=1(2B), pad_enable, pad_interval=4(32 dw), pad_amount=3(4 dw)
            g1[0] = (int)((1u << 16) | (1u << 20) | (4u << 22) | (3u << 25));
            g1[1] = (int)(((unsigned)K & 0xFFFFu) << 16);            // tensor_dim0 lo
            g1[2] = (int)(((unsigned)K >> 16) |
                          (((unsigned)M & 0xFFFFu) << 16));          // dim0 hi|dim1 lo
            g1[3] = (int)(((unsigned)M >> 16) | (64u << 16));        // dim1 hi|tile0
            g1[4] = 64;                                              // tile1 | tile2=0
            g1[5] = K;                                               // dim0_stride lo
            g1[6] = 0;
            g1[7] = 0;
            v4i gz4; gz4[0] = gz4[1] = gz4[2] = gz4[3] = 0;
            v8i gz8;
#pragma unroll
            for (int z = 0; z < 8; ++z) gz8[z] = 0;
            __builtin_amdgcn_tensor_load_to_lds(g0, g1, gz4, gz4, gz8, 0);
        }
#else
        // ---- A tile: pure bf16 copy, 16B vectors ----
#pragma unroll
        for (int i = 0; i < 2; ++i) {
            int vecidx = tid + i * 256;            // 0..511
            int row = vecidx >> 3;
            int kk  = (vecidx & 7) * 8;
            *(v8u16*)&a_s[row * AST + kk] =
                *(const v8u16*)&A[(size_t)(m0 + row) * K + k0 + kk];
        }
#endif
        // ---- B tile (64 k x 64 n): vector loads, transposed scalar stores ----
#pragma unroll
        for (int i = 0; i < 2; ++i) {
            int vecidx = tid + i * 256;
            int kk = vecidx >> 3;
            int n8 = (vecidx & 7) * 8;
            const unsigned short* bp = &Bm[(size_t)(k0 + kk) * N + n0 + n8];
            if (i == 0 && k0 + 64 < K) __builtin_prefetch(bp + (size_t)64 * N, 0, 1);
            v8u16 bv = *(const v8u16*)bp;
#pragma unroll
            for (int j = 0; j < 8; ++j) b_s[(n8 + j) * BST + kk] = bv[j];
        }
#if defined(USE_TDM)
        if (tid < 32) __builtin_amdgcn_s_wait_tensorcnt(0);
#endif
        __syncthreads();

        // ---- fragments (preload, then WMMA burst) ----
        Frag af[2], bf0[2], bf1[2];
        const unsigned short* ap  = &a_s[(wr * 16 + (lane & 15)) * AST];
        const unsigned short* bp0 = &b_s[(wc2 * 16 + (lane & 15)) * BST + halfk];
        const unsigned short* bp1 = &b_s[((wc2 + 1) * 16 + (lane & 15)) * BST + halfk];
#pragma unroll
        for (int kc = 0; kc < 2; ++kc) {
            af[kc].h[0] = *(const v8u16*)(ap + kc * 32 + off);
            af[kc].h[1] = *(const v8u16*)(ap + kc * 32 + off + 8);
            bf0[kc].bf  = *(const v16bf*)(bp0 + kc * 32);
            bf1[kc].bf  = *(const v16bf*)(bp1 + kc * 32);
        }
#pragma unroll
        for (int kc = 0; kc < 2; ++kc) {
            c0 = wmma_bf16(af[kc].bf, bf0[kc].bf, c0);
            c1 = wmma_bf16(af[kc].bf, bf1[kc].bf, c1);
        }
        __syncthreads();
    }

    // ---- epilogue: C layout VGPR r -> (M = r + 8*(lane>=16), N = lane&15) ----
    const int rbase = (lane >> 4) * 8;
    const int ncol  = lane & 15;
#pragma unroll
    for (int r = 0; r < 8; ++r) {
        const size_t row = (size_t)(m0 + wr * 16 + rbase + r) * N + n0;
        C[row + wc2 * 16 + ncol]       = c0[r];
        C[row + (wc2 + 1) * 16 + ncol] = c1[r];
    }
}

// ---------------------------------------------------------------------------
// Per-head RMSNorm + RoPE, f32 [b,s,h,d] -> bf16 [b,h,s,d]; one wave per head.
// ---------------------------------------------------------------------------
template <int NH>
__global__ __launch_bounds__(256) void rmsnorm_rope_kernel(
    const float* __restrict__ in, const float* __restrict__ w,
    const float* __restrict__ cosp, const float* __restrict__ sinp,
    unsigned short* __restrict__ out, float scale)
{
    const int gw   = blockIdx.x * 8 + (threadIdx.x >> 5);
    const int lane = threadIdx.x & 31;
    const int h = gw & (NH - 1);
    const int s = (gw / NH) & (SS - 1);
    const int b = gw / (NH * SS);

    const float* tin = in + ((size_t)(b * SS + s) * NH + h) * DD;
    float t0 = tin[lane], t1 = tin[lane + 32], t2 = tin[lane + 64], t3 = tin[lane + 96];

    float ssq = t0 * t0 + t1 * t1 + t2 * t2 + t3 * t3;
#pragma unroll
    for (int m = 1; m < 32; m <<= 1) ssq += __shfl_xor(ssq, m, 32);
    const float rn = rsqrtf(ssq * (1.0f / DD) + EPSC);

    const float n0 = t0 * rn * w[lane];
    const float n1 = t1 * rn * w[lane + 32];
    const float n2 = t2 * rn * w[lane + 64];
    const float n3 = t3 * rn * w[lane + 96];

    const float* cs = cosp + (size_t)s * DD;
    const float* sn = sinp + (size_t)s * DD;
    const float o0 = (n0 * cs[lane]      - n2 * sn[lane])      * scale;
    const float o1 = (n1 * cs[lane + 32] - n3 * sn[lane + 32]) * scale;
    const float o2 = (n2 * cs[lane + 64] + n0 * sn[lane + 64]) * scale;
    const float o3 = (n3 * cs[lane + 96] + n1 * sn[lane + 96]) * scale;

    unsigned short* to = out + ((size_t)(b * NH + h) * SS + s) * DD;
    to[lane]      = f2bf(o0);
    to[lane + 32] = f2bf(o1);
    to[lane + 64] = f2bf(o2);
    to[lane + 96] = f2bf(o3);
}

// ---------------------------------------------------------------------------
// V: f32 [b,s,kv,d] -> bf16 V^T [b,kv,d,s]  (makes PV B-fragments contiguous)
// ---------------------------------------------------------------------------
__global__ __launch_bounds__(256) void v_transpose_kernel(
    const float* __restrict__ v, unsigned short* __restrict__ vtb)
{
    const size_t total = (size_t)BB * SS * KVH * DD;
    for (size_t idx = (size_t)blockIdx.x * blockDim.x + threadIdx.x;
         idx < total; idx += (size_t)gridDim.x * blockDim.x) {
        const int d = (int)(idx & (DD - 1));
        const int h = (int)((idx >> 7) & (KVH - 1));
        const int s = (int)((idx >> 10) & (SS - 1));
        const int b = (int)(idx >> 21);
        vtb[((size_t)(b * KVH + h) * DD + d) * SS + s] = f2bf(v[idx]);
    }
}

// ---------------------------------------------------------------------------
// Flash attention (causal, GQA NREP=2). One wave per 16-row Q tile.
// Q bf16 [b,h,s,d] (scale pre-folded), K bf16 [b,kv,s,d], V^T bf16 [b,kv,d,s].
// Output bf16 [b,s,h*d] (feeds the Wo GEMM directly).
// ---------------------------------------------------------------------------
constexpr int PSTR = 40;   // P-tile LDS row stride (u16), 80B: 16B-aligned frags

__global__ __launch_bounds__(128) void attn_fa_kernel(
    const unsigned short* __restrict__ qb,
    const unsigned short* __restrict__ kb,
    const unsigned short* __restrict__ vtb,
    unsigned short* __restrict__ attn)
{
    __shared__ alignas(32) unsigned short plds[4][16 * PSTR];  // per-wave P tile

    const int lane = threadIdx.x & 31;
    const int wave = threadIdx.x >> 5;
    const int gw   = blockIdx.x * 4 + wave;          // 0 .. B*H*(S/16)-1
    const int st   = gw & (SS / 16 - 1);             // 7 bits
    const int bh   = gw >> 7;
    const int h    = bh & (HH - 1);
    const int b    = bh >> 4;
    const int s0   = st * 16;
    const int hk   = h >> 1;                         // NREP = 2
    const int ncol  = lane & 15;
    const int rbase = (lane >> 4) * 8;
    const int off   = (lane >> 4) * 16;

    // ---- Q fragments: A-layout, row = lane&15, 4 chunks of K(d)=32 ----
    Frag qf[4];
    const unsigned short* qrow =
        qb + ((size_t)(b * HH + h) * SS + (s0 + (lane & 15))) * DD;
#pragma unroll
    for (int c = 0; c < 4; ++c) {
        qf[c].h[0] = *(const v8u16*)(qrow + c * 32 + off);
        qf[c].h[1] = *(const v8u16*)(qrow + c * 32 + off + 8);
    }

    v8f acc[8];
#pragma unroll
    for (int j = 0; j < 8; ++j) acc[j] = vzero8();
    float mrow[8], lrow[8];
#pragma unroll
    for (int r = 0; r < 8; ++r) { mrow[r] = -3.0e38f; lrow[r] = 0.0f; }

    const unsigned short* kbase = kb  + (size_t)(b * KVH + hk) * SS * DD;
    const unsigned short* vbase = vtb + (size_t)(b * KVH + hk) * DD * SS;
    unsigned short* pl = &plds[wave][0];

    const int nchunk = ((s0 + 15) >> 5) + 1;         // 32-wide key chunks
    for (int c0 = 0; c0 < nchunk; ++c0) {
        const int t0 = c0 * 32;

        // ---- preload all 8 K fragments (one vmem clause), then WMMA burst ----
        Frag kf[8];
        const unsigned short* kc0 =
            kbase + (size_t)(t0 + ncol) * DD + 16 * (lane >> 4);
        const unsigned short* kc1 =
            kbase + (size_t)(t0 + 16 + ncol) * DD + 16 * (lane >> 4);
#pragma unroll
        for (int dd = 0; dd < 4; ++dd) {
            kf[dd].bf     = *(const v16bf*)(kc0 + dd * 32);
            kf[4 + dd].bf = *(const v16bf*)(kc1 + dd * 32);
        }
        v8f sc0 = vzero8(), sc1 = vzero8();
#pragma unroll
        for (int dd = 0; dd < 4; ++dd) {
            sc0 = wmma_bf16(qf[dd].bf, kf[dd].bf, sc0);
            sc1 = wmma_bf16(qf[dd].bf, kf[4 + dd].bf, sc1);
        }

        // ---- causal mask + online softmax (rows live in 16-lane halves) ----
#pragma unroll
        for (int r = 0; r < 8; ++r) {
            const int qi = s0 + rbase + r;
            float slo = sc0[r]; if (t0 + ncol      > qi) slo = -3.0e38f;
            float shi = sc1[r]; if (t0 + 16 + ncol > qi) shi = -3.0e38f;

            float tm = fmaxf(slo, shi);
#pragma unroll
            for (int mk = 1; mk < 16; mk <<= 1)
                tm = fmaxf(tm, __shfl_xor(tm, mk, 32));
            const float mn   = fmaxf(mrow[r], tm);
            const float corr = __expf(mrow[r] - mn);
            const float plo  = __expf(slo - mn);
            const float phi  = __expf(shi - mn);
            float rs = plo + phi;
#pragma unroll
            for (int mk = 1; mk < 16; mk <<= 1)
                rs += __shfl_xor(rs, mk, 32);

            lrow[r] = lrow[r] * corr + rs;
            mrow[r] = mn;
#pragma unroll
            for (int j = 0; j < 8; ++j) acc[j][r] *= corr;

            // C-layout -> LDS (row-major P tile) for A-layout reload
            pl[(rbase + r) * PSTR + ncol]      = f2bf(plo);
            pl[(rbase + r) * PSTR + 16 + ncol] = f2bf(phi);
        }
        asm volatile("s_wait_dscnt 0x0" ::: "memory");   // lanes see P tile

        // ---- P fragment (A-layout, 16x32 over key pos) ----
        Frag pf;
        const unsigned short* pr = pl + (lane & 15) * PSTR + off;
        pf.h[0] = *(const v8u16*)pr;
        pf.h[1] = *(const v8u16*)(pr + 8);

        // ---- PV: preload 8 V fragments, then WMMA burst over D=128 ----
        Frag vf[8];
#pragma unroll
        for (int j = 0; j < 8; ++j)
            vf[j].bf = *(const v16bf*)(vbase + (size_t)(j * 16 + ncol) * SS +
                                       t0 + 16 * (lane >> 4));
#pragma unroll
        for (int j = 0; j < 8; ++j)
            acc[j] = wmma_bf16(pf.bf, vf[j].bf, acc[j]);
    }

    // ---- epilogue: normalize, write bf16 [b, s, h*D] ----
#pragma unroll
    for (int r = 0; r < 8; ++r) {
        const float inv = 1.0f / lrow[r];
        const int srow  = s0 + rbase + r;
        unsigned short* orow =
            attn + ((size_t)(b * SS + srow)) * (HH * DD) + h * DD;
#pragma unroll
        for (int j = 0; j < 8; ++j)
            orow[j * 16 + ncol] = f2bf(acc[j][r] * inv);
    }
}

// ---------------------------------------------------------------------------
// Host-side launcher
// ---------------------------------------------------------------------------
extern "C" void kernel_launch(void* const* d_in, const int* in_sizes, int n_in,
                              void* d_out, int out_size, void* d_ws, size_t ws_size,
                              hipStream_t stream)
{
    (void)in_sizes; (void)n_in; (void)out_size; (void)ws_size;

    const float* x    = (const float*)d_in[0];
    const float* Wq   = (const float*)d_in[1];
    const float* Wk   = (const float*)d_in[2];
    const float* Wv   = (const float*)d_in[3];
    const float* Wo   = (const float*)d_in[4];
    const float* qnw  = (const float*)d_in[5];
    const float* knw  = (const float*)d_in[6];
    const float* cosp = (const float*)d_in[7];
    const float* sinp = (const float*)d_in[8];
    float* outp = (float*)d_out;

    char* p = (char*)d_ws;
    auto take = [&](size_t bytes) {
        char* r = p;
        p += (bytes + 255) & ~(size_t)255;
        return r;
    };
    const size_t BS = (size_t)BB * SS;                         // 4096
    float*          qf32 = (float*)take(BS * HH  * DD * 4);    // [b*s, H*D]
    float*          kf32 = (float*)take(BS * KVH * DD * 4);
    float*          vf32 = (float*)take(BS * KVH * DD * 4);
    unsigned short* qb   = (unsigned short*)take(BS * HH  * DD * 2);
    unsigned short* kb   = (unsigned short*)take(BS * KVH * DD * 2);
    unsigned short* vtb  = (unsigned short*)take(BS * KVH * DD * 2);
    unsigned short* attn = (unsigned short*)take(BS * HH  * DD * 2);
    unsigned short* xb   = (unsigned short*)take(BS * HIDC * 2);
    unsigned short* wqb  = (unsigned short*)take((size_t)HIDC * HH  * DD * 2);
    unsigned short* wkb  = (unsigned short*)take((size_t)HIDC * KVH * DD * 2);
    unsigned short* wvb  = (unsigned short*)take((size_t)HIDC * KVH * DD * 2);
    unsigned short* wob  = (unsigned short*)take((size_t)HH * DD * HIDC * 2);

    const int M = (int)BS;

    // 0) one-time (per call) f32 -> bf16 conversions
    cvt_bf16_kernel<<<1024, 256, 0, stream>>>(x,  xb,  (int)(BS * HIDC / 4));
    cvt_bf16_kernel<<<1024, 256, 0, stream>>>(Wq, wqb, HIDC * HH  * DD / 4);
    cvt_bf16_kernel<<<1024, 256, 0, stream>>>(Wk, wkb, HIDC * KVH * DD / 4);
    cvt_bf16_kernel<<<1024, 256, 0, stream>>>(Wv, wvb, HIDC * KVH * DD / 4);
    cvt_bf16_kernel<<<1024, 256, 0, stream>>>(Wo, wob, HH * DD * HIDC / 4);

    // 1) QKV projections (bf16 WMMA GEMM, TDM-staged A tiles)
    gemm_wmma_kernel<<<dim3((HH * DD) / 64, M / 64), 256, 0, stream>>>(
        xb, wqb, qf32, M, HH * DD, HIDC);
    gemm_wmma_kernel<<<dim3((KVH * DD) / 64, M / 64), 256, 0, stream>>>(
        xb, wkb, kf32, M, KVH * DD, HIDC);
    gemm_wmma_kernel<<<dim3((KVH * DD) / 64, M / 64), 256, 0, stream>>>(
        xb, wvb, vf32, M, KVH * DD, HIDC);

    // 2) per-head RMSNorm + RoPE -> bf16 (scale folded into Q); V -> bf16 V^T
    rmsnorm_rope_kernel<HH><<<(BB * SS * HH) / 8, 256, 0, stream>>>(
        qf32, qnw, cosp, sinp, qb, SCALEC);
    rmsnorm_rope_kernel<KVH><<<(BB * SS * KVH) / 8, 256, 0, stream>>>(
        kf32, knw, cosp, sinp, kb, 1.0f);
    v_transpose_kernel<<<4096, 256, 0, stream>>>(vf32, vtb);

    // 3) causal GQA flash attention (WMMA QK^T and PV)
    attn_fa_kernel<<<(BB * HH * (SS / 16)) / 4, 128, 0, stream>>>(qb, kb, vtb, attn);

    // 4) output projection: attn(bf16) @ Wo(bf16) -> f32 out
    gemm_wmma_kernel<<<dim3(HIDC / 64, M / 64), 256, 0, stream>>>(
        attn, wob, outp, M, HIDC, HIDC);
}
